// LSTMDecoder_18133351924313
// MI455X (gfx1250) — compile-verified, matrix operand order checked by reading, not run on hardware
//
#include <hip/hip_runtime.h>
#include <stdint.h>

typedef __bf16 bf16;
typedef __attribute__((ext_vector_type(16))) __bf16 bf16x16;
typedef __attribute__((ext_vector_type(8)))  __bf16 bf16x8;
typedef __attribute__((ext_vector_type(8)))  float  f32x8;

#define VSZ   10000
#define ESZ   512
#define HSZ   512
#define ENCSZ 2048
#define BSZ   64
#define TSZ   64
#define PSZ   196
#define G4H   2048   // 4*H
#define TS    63     // T-1 recurrence steps

// ---------------------------------------------------------------------------
// Load a 16x32 bf16 tile (rows row0..row0+15, cols k0..k0+31) from a row-major
// [rows][ld] matrix into the CDNA5 WMMA A/B register layout:
//   lane l (l<16):  row=row0+l,  elems 0..7 -> K=k0+half*8+[0..8),
//                                elems 8..15 -> K=k0+16+half*8+[0..8)
//   where half = lane>>4. Both A (M x K) and B-as-(N x K) use this layout.
// ---------------------------------------------------------------------------
__device__ __forceinline__ bf16x16 load_tile16x32(const bf16* __restrict__ base,
                                                  int ld, int row0, int k0) {
    const int lane = threadIdx.x & 31;
    const int r    = row0 + (lane & 15);
    const int half = (lane >> 4) & 1;
    const bf16* p  = base + (size_t)r * ld + k0 + half * 8;
    union { bf16x16 v; bf16x8 h[2]; } u;
    u.h[0] = *(const bf16x8*)(p);        // 16B aligned (ld, k0 multiples of 8)
    u.h[1] = *(const bf16x8*)(p + 16);
    return u.v;
}

__device__ __forceinline__ float sigf(float x) {
    return 1.0f / (1.0f + __expf(-x));
}

// ---------------------------------------------------------------------------
// 1) mean over P, emit bf16 mean_enc (64 x 2048)
// ---------------------------------------------------------------------------
__global__ void k_mean_enc(const float* __restrict__ enc, bf16* __restrict__ mean_bf) {
    int idx = blockIdx.x * blockDim.x + threadIdx.x;   // 64*2048 threads
    int b = idx >> 11, e = idx & 2047;
    const float* p = enc + ((size_t)b * PSZ) * ENCSZ + e;
    float s = 0.f;
    for (int j = 0; j < PSZ; ++j) s += p[(size_t)j * ENCSZ];
    mean_bf[(size_t)b * ENCSZ + e] = (bf16)(s * (1.0f / (float)PSZ));
}

// ---------------------------------------------------------------------------
// 2) generic fp32 -> bf16 weight conversion
// ---------------------------------------------------------------------------
__global__ void k_f32_to_bf16(const float* __restrict__ src, bf16* __restrict__ dst, int n) {
    int i = blockIdx.x * blockDim.x + threadIdx.x;
    if (i < n) dst[i] = (bf16)src[i];
}

// ---------------------------------------------------------------------------
// 3) embedding gather -> emb_bf laid out [t][b][e] (t = 0..62)
// ---------------------------------------------------------------------------
__global__ void k_gather_emb(const int* __restrict__ targets,
                             const float* __restrict__ embedding,
                             bf16* __restrict__ emb_bf) {
    int idx = blockIdx.x * blockDim.x + threadIdx.x;   // 63*64*512
    if (idx >= TS * BSZ * ESZ) return;
    int e  = idx & 511;
    int bt = idx >> 9;          // t*64 + b
    int b  = bt & 63;
    int t  = bt >> 6;
    int tok = targets[b * TSZ + t];                    // targets[:, :-1]
    emb_bf[idx] = (bf16)embedding[(size_t)tok * ESZ + e];
}

// ---------------------------------------------------------------------------
// 4) h0/c0 = tanh(mean_enc @ W.T + b)   (M=64, N=512, K=2048), z=0:h, z=1:c
// ---------------------------------------------------------------------------
__global__ void k_init_state(const bf16* __restrict__ mean_bf,
                             const bf16* __restrict__ Wh_bf, const float* __restrict__ bh,
                             const bf16* __restrict__ Wc_bf, const float* __restrict__ bc,
                             bf16* __restrict__ h_bf, float* __restrict__ c) {
    const int tn = blockIdx.x;              // 0..31
    const int tm = blockIdx.y;              // 0..3
    const bool is_c = (blockIdx.z != 0);
    const bf16*  W    = is_c ? Wc_bf : Wh_bf;
    const float* bias = is_c ? bc : bh;
    f32x8 acc = {};
    #pragma unroll 4
    for (int k = 0; k < ENCSZ; k += 32) {
        bf16x16 a = load_tile16x32(mean_bf, ENCSZ, tm * 16, k);
        bf16x16 w = load_tile16x32(W,       ENCSZ, tn * 16, k);
        acc = __builtin_amdgcn_wmma_f32_16x16x32_bf16(false, a, false, w,
                                                      (short)0, acc, false, false);
    }
    const int lane = threadIdx.x & 31;
    const int n  = tn * 16 + (lane & 15);
    const int m0 = tm * 16 + ((lane >> 4) ? 8 : 0);
    const float bn = bias[n];
    #pragma unroll
    for (int r = 0; r < 8; ++r) {
        float v = tanhf(acc[r] + bn);
        int o = (m0 + r) * HSZ + n;
        if (is_c) c[o] = v; else h_bf[o] = (bf16)v;
    }
}

// ---------------------------------------------------------------------------
// 5) xg = emb @ W_ih.T + (b_ih + b_hh)   (M=4032, N=2048, K=512), fp32 out
// ---------------------------------------------------------------------------
__global__ void k_xg_gemm(const bf16* __restrict__ emb_bf, const bf16* __restrict__ Wih_bf,
                          const float* __restrict__ b_ih, const float* __restrict__ b_hh,
                          float* __restrict__ xg) {
    const int tn = blockIdx.x;              // 0..127
    const int tm = blockIdx.y;              // 0..251
    f32x8 acc = {};
    #pragma unroll 4
    for (int k = 0; k < ESZ; k += 32) {
        bf16x16 a = load_tile16x32(emb_bf, ESZ, tm * 16, k);
        bf16x16 w = load_tile16x32(Wih_bf, ESZ, tn * 16, k);
        acc = __builtin_amdgcn_wmma_f32_16x16x32_bf16(false, a, false, w,
                                                      (short)0, acc, false, false);
    }
    const int lane = threadIdx.x & 31;
    const int n  = tn * 16 + (lane & 15);
    const int m0 = tm * 16 + ((lane >> 4) ? 8 : 0);
    const float bn = b_ih[n] + b_hh[n];
    #pragma unroll
    for (int r = 0; r < 8; ++r)
        xg[(size_t)(m0 + r) * G4H + n] = acc[r] + bn;
}

// ---------------------------------------------------------------------------
// 6) per-step: gates = xg[t] + h @ W_hh.T   (M=64, N=2048, K=512)
// ---------------------------------------------------------------------------
__global__ void k_gates_gemm(const bf16* __restrict__ h_bf, const bf16* __restrict__ Whh_bf,
                             const float* __restrict__ xg_t, float* __restrict__ gates) {
    const int tn = blockIdx.x;              // 0..127
    const int tm = blockIdx.y;              // 0..3
    f32x8 acc = {};
    #pragma unroll 4
    for (int k = 0; k < HSZ; k += 32) {
        bf16x16 a = load_tile16x32(h_bf,   HSZ, tm * 16, k);
        bf16x16 w = load_tile16x32(Whh_bf, HSZ, tn * 16, k);
        acc = __builtin_amdgcn_wmma_f32_16x16x32_bf16(false, a, false, w,
                                                      (short)0, acc, false, false);
    }
    const int lane = threadIdx.x & 31;
    const int n  = tn * 16 + (lane & 15);
    const int m0 = tm * 16 + ((lane >> 4) ? 8 : 0);
    #pragma unroll
    for (int r = 0; r < 8; ++r) {
        size_t o = (size_t)(m0 + r) * G4H + n;
        gates[o] = acc[r] + xg_t[o];
    }
}

// ---------------------------------------------------------------------------
// 7) per-step: LSTM cell elementwise; updates c (fp32) and h (bf16)
// ---------------------------------------------------------------------------
__global__ void k_cell(const float* __restrict__ gates, float* __restrict__ c,
                       bf16* __restrict__ h_bf) {
    int idx = blockIdx.x * blockDim.x + threadIdx.x;   // 64*512
    int b = idx >> 9, j = idx & 511;
    const float* g = gates + (size_t)b * G4H;
    float ig = sigf(g[j]);
    float fg = sigf(g[512 + j]);
    float gg = tanhf(g[1024 + j]);
    float og = sigf(g[1536 + j]);
    float cn = fg * c[idx] + ig * gg;
    c[idx] = cn;
    h_bf[idx] = (bf16)(og * tanhf(cn));
}

// ---------------------------------------------------------------------------
// 8) per-step: out[:, t+1, :] = h @ Wout.T + bout  (M=64, N=10000, K=512)
// ---------------------------------------------------------------------------
__global__ void k_proj(const bf16* __restrict__ h_bf, const bf16* __restrict__ Wout_bf,
                       const float* __restrict__ bout, float* __restrict__ out, int t1) {
    const int tn = blockIdx.x;              // 0..624  (625*16 == 10000)
    const int tm = blockIdx.y;              // 0..3
    f32x8 acc = {};
    #pragma unroll 4
    for (int k = 0; k < HSZ; k += 32) {
        bf16x16 a = load_tile16x32(h_bf,    HSZ, tm * 16, k);
        bf16x16 w = load_tile16x32(Wout_bf, HSZ, tn * 16, k);
        acc = __builtin_amdgcn_wmma_f32_16x16x32_bf16(false, a, false, w,
                                                      (short)0, acc, false, false);
    }
    const int lane = threadIdx.x & 31;
    const int n  = tn * 16 + (lane & 15);
    const int m0 = tm * 16 + ((lane >> 4) ? 8 : 0);
    const float bn = bout[n];
    #pragma unroll
    for (int r = 0; r < 8; ++r)
        out[((size_t)(m0 + r) * TSZ + t1) * VSZ + n] = acc[r] + bn;
}

// ---------------------------------------------------------------------------
// 9) zero row t=0 of the output
// ---------------------------------------------------------------------------
__global__ void k_zero_t0(float* __restrict__ out) {
    int idx = blockIdx.x * blockDim.x + threadIdx.x;   // 64*10000
    if (idx < BSZ * VSZ) {
        int b = idx / VSZ, v = idx % VSZ;
        out[(size_t)b * TSZ * VSZ + v] = 0.f;
    }
}

// ---------------------------------------------------------------------------
extern "C" void kernel_launch(void* const* d_in, const int* in_sizes, int n_in,
                              void* d_out, int out_size, void* d_ws, size_t ws_size,
                              hipStream_t stream) {
    const float* enc       = (const float*)d_in[0];
    const int*   targets   = (const int*)  d_in[1];
    const float* embedding = (const float*)d_in[2];
    const float* W_ih      = (const float*)d_in[3];
    const float* W_hh      = (const float*)d_in[4];
    const float* b_ih      = (const float*)d_in[5];
    const float* b_hh      = (const float*)d_in[6];
    const float* Wh        = (const float*)d_in[7];
    const float* bh        = (const float*)d_in[8];
    const float* Wc        = (const float*)d_in[9];
    const float* bc        = (const float*)d_in[10];
    const float* Wout      = (const float*)d_in[11];
    const float* bout      = (const float*)d_in[12];
    float* out = (float*)d_out;

    uint8_t* ws = (uint8_t*)d_ws;
    size_t off = 0;
    auto carve = [&](size_t bytes) -> uint8_t* {
        uint8_t* p = ws + off;
        off += (bytes + 255) & ~(size_t)255;
        return p;
    };
    bf16*  mean_bf = (bf16*) carve((size_t)BSZ * ENCSZ * 2);
    bf16*  Wih_bf  = (bf16*) carve((size_t)G4H * ESZ   * 2);
    bf16*  Whh_bf  = (bf16*) carve((size_t)G4H * HSZ   * 2);
    bf16*  Wh_bf   = (bf16*) carve((size_t)HSZ * ENCSZ * 2);
    bf16*  Wc_bf   = (bf16*) carve((size_t)HSZ * ENCSZ * 2);
    bf16*  Wout_bf = (bf16*) carve((size_t)VSZ * HSZ   * 2);
    bf16*  emb_bf  = (bf16*) carve((size_t)TS * BSZ * ESZ * 2);
    float* xg      = (float*)carve((size_t)TS * BSZ * G4H * 4);
    float* gates   = (float*)carve((size_t)BSZ * G4H * 4);
    bf16*  h_bf    = (bf16*) carve((size_t)BSZ * HSZ * 2);
    float* cst     = (float*)carve((size_t)BSZ * HSZ * 4);
    (void)ws_size; (void)in_sizes; (void)n_in; (void)out_size;

    // --- parallel preamble -------------------------------------------------
    k_mean_enc<<<(BSZ * ENCSZ) / 256, 256, 0, stream>>>(enc, mean_bf);

    k_f32_to_bf16<<<(G4H * ESZ + 255) / 256, 256, 0, stream>>>(W_ih, Wih_bf, G4H * ESZ);
    k_f32_to_bf16<<<(G4H * HSZ + 255) / 256, 256, 0, stream>>>(W_hh, Whh_bf, G4H * HSZ);
    k_f32_to_bf16<<<(HSZ * ENCSZ + 255) / 256, 256, 0, stream>>>(Wh, Wh_bf, HSZ * ENCSZ);
    k_f32_to_bf16<<<(HSZ * ENCSZ + 255) / 256, 256, 0, stream>>>(Wc, Wc_bf, HSZ * ENCSZ);
    k_f32_to_bf16<<<(VSZ * HSZ + 255) / 256, 256, 0, stream>>>(Wout, Wout_bf, VSZ * HSZ);

    k_gather_emb<<<(TS * BSZ * ESZ + 255) / 256, 256, 0, stream>>>(targets, embedding, emb_bf);

    k_init_state<<<dim3(HSZ / 16, BSZ / 16, 2), 32, 0, stream>>>(mean_bf, Wh_bf, bh,
                                                                 Wc_bf, bc, h_bf, cst);
    k_xg_gemm<<<dim3(G4H / 16, (TS * BSZ) / 16), 32, 0, stream>>>(emb_bf, Wih_bf,
                                                                  b_ih, b_hh, xg);
    k_zero_t0<<<(BSZ * VSZ + 255) / 256, 256, 0, stream>>>(out);

    // --- serial recurrence: 63 steps --------------------------------------
    for (int t = 0; t < TS; ++t) {
        k_gates_gemm<<<dim3(G4H / 16, BSZ / 16), 32, 0, stream>>>(
            h_bf, Whh_bf, xg + (size_t)t * BSZ * G4H, gates);
        k_cell<<<(BSZ * HSZ) / 256, 256, 0, stream>>>(gates, cst, h_bf);
        k_proj<<<dim3(VSZ / 16, BSZ / 16), 32, 0, stream>>>(h_bf, Wout_bf, bout, out, t + 1);
    }
}
